// PclCldLoss_2_13761075216975
// MI455X (gfx1250) — compile-verified, matrix operand order checked by reading, not run on hardware
//
#include <hip/hip_runtime.h>
#include <hip/hip_bf16.h>

typedef __attribute__((ext_vector_type(16))) _Float16 v16h;
typedef __attribute__((ext_vector_type(8)))  _Float16 v8h;
typedef __attribute__((ext_vector_type(4)))  _Float16 v4h;
typedef __attribute__((ext_vector_type(8)))  float    v8f;
typedef __attribute__((ext_vector_type(4)))  float    v4f;
typedef __attribute__((ext_vector_type(4)))  int      v4i;

#define BSZ 16384
#define CSZ 2048
#define DSZ 128
#define ROWP 136            // padded row stride in halfs (272 B = 17*16 B -> conflict-free b128)

// ---------------------------------------------------------------------------
// async 16-byte global->LDS copy (CDNA5 GLOBAL_LOAD_ASYNC_TO_LDS_B128),
// with a synchronous register-copy fallback if the builtin is unavailable.
// ---------------------------------------------------------------------------
__device__ __forceinline__ void async_ld16(_Float16* lds, const _Float16* g)
{
#if __has_builtin(__builtin_amdgcn_global_load_async_to_lds_b128)
    __builtin_amdgcn_global_load_async_to_lds_b128(
        (__attribute__((address_space(1))) v4i*)(g),
        (__attribute__((address_space(3))) v4i*)(lds),
        0, 0);
#else
    *(v8h*)lds = *(const v8h*)g;
#endif
}

__device__ __forceinline__ void wait_async0()
{
#if __has_builtin(__builtin_amdgcn_s_wait_asynccnt)
    __builtin_amdgcn_s_wait_asynccnt(0);
#else
    asm volatile("s_wait_asynccnt 0x0" ::: "memory");
#endif
}

// ---------------------------------------------------------------------------
// prep: convert centroid matrices f32 -> f16 into workspace, invert conc.
// grid: 1024 x 256  (262144 threads == C*D)
// ---------------------------------------------------------------------------
__global__ __launch_bounds__(256) void pcl_prep(
    const float* __restrict__ M,  const float* __restrict__ MI,
    const float* __restrict__ c,  const float* __restrict__ cI,
    _Float16* __restrict__ M16,   _Float16* __restrict__ M16I,
    float* __restrict__ ivc,      float* __restrict__ ivcI)
{
    int idx = blockIdx.x * 256 + threadIdx.x;
    if (idx < CSZ * DSZ) {
        M16[idx]  = (_Float16)M[idx];
        M16I[idx] = (_Float16)MI[idx];
    }
    if (idx < CSZ) {
        ivc[idx]  = 1.0f / c[idx];
        ivcI[idx] = 1.0f / cI[idx];
    }
}

// ---------------------------------------------------------------------------
// main: fused GEMM + exp-reduction + positive gather.
// grid: 256 workgroups (side = bid>>7, column-block = bid&127), 256 threads.
// A (centroids) streamed through a 2-deep async LDS ring; B (features) held
// in registers per wave; 4 chained v_wmma_f32_16x16x32_f16 per 16-row chunk.
// ---------------------------------------------------------------------------
__global__ __launch_bounds__(256) void pcl_main(
    const float* __restrict__ features,
    const float* __restrict__ features_I,
    const _Float16* __restrict__ M16,
    const _Float16* __restrict__ M16_I,
    const float* __restrict__ invc,
    const float* __restrict__ invc_I,
    const int*  __restrict__ labels,
    const int*  __restrict__ labels_I,
    float* __restrict__ partials)
{
    __shared__ _Float16 sB[128 * ROWP];     // 34 KB  feature subtile (f16, padded)
    __shared__ _Float16 sA[2][16 * ROWP];   // 8.5 KB centroid chunk ring (padded)
    __shared__ float    sIC[CSZ];           // 8 KB   1/concentration for this side
    __shared__ float    red[8];

    const int tid  = threadIdx.x;
    const int side = blockIdx.x >> 7;
    const int cb   = blockIdx.x & 127;
    const int b0   = cb * 128;

    // side 0: logits   = M_kmeans   x features_I^T  (conc,   labels)
    // side 1: logits_I = M_kmeans_I x features^T    (conc_I, labels_I)
    const float*    feat = side ? features : features_I;
    const _Float16* Amat = side ? M16_I    : M16;
    const float*    icg  = side ? invc_I   : invc;
    const int*      lab  = side ? labels_I : labels;

    // ---- stage 128x128 feature subtile into LDS (f32->f16, padded rows) ----
    const float* fsrc = feat + (size_t)b0 * DSZ;
    for (int e4 = tid; e4 < (128 * DSZ) / 4; e4 += 256) {
        int e = e4 * 4;
        int r = e >> 7, cc = e & 127;
        v4f f = *(const v4f*)(fsrc + (size_t)e);
        v4h h;
        h[0] = (_Float16)f[0]; h[1] = (_Float16)f[1];
        h[2] = (_Float16)f[2]; h[3] = (_Float16)f[3];
        *(v4h*)(&sB[r * ROWP + cc]) = h;
    }
    // ---- stage 1/concentration vector into LDS ----
    for (int idx = tid; idx < CSZ; idx += 256)
        sIC[idx] = icg[idx];
    __syncthreads();

    const int lane = tid & 31;
    const int wave = tid >> 5;
    const int colL = wave * 16 + (lane & 15);   // local column (0..127)
    const int sel  = lane >> 4;                 // lane half: 0 or 1
    const int lbl  = lab[b0 + colL];
    const int rel  = lbl - sel * 8;             // rebased label for compares

    // ---- B operand registers for all 4 K-chunks (held for entire loop) ----
    v16h breg[4];
    #pragma unroll
    for (int kc = 0; kc < 4; ++kc) {
        const _Float16* bp = &sB[colL * ROWP + kc * 32 + sel * 8];
        v8h lo = *(const v8h*)(bp);
        v8h hi = *(const v8h*)(bp + 16);
        breg[kc] = __builtin_shufflevector(lo, hi,
            0, 1, 2, 3, 4, 5, 6, 7, 8, 9, 10, 11, 12, 13, 14, 15);
    }

    const int arow = lane & 15;   // A: lanes m and m+16 both hold row m
    const int koff = sel * 8;
    const int srow = tid >> 4;    // cooperative staging: row (0..15)
    const int sseg = tid & 15;    // 16-byte segment within row

    // prefetch chunk 0 into ring slot 0
    async_ld16(&sA[0][srow * ROWP + sseg * 8],
               Amat + (size_t)srow * DSZ + sseg * 8);

    float sumexp = 0.0f;
    float pos    = 0.0f;

    for (int i = 0; i < CSZ / 16; ++i) {
        wait_async0();                 // own part of chunk i arrived
        __syncthreads();               // whole chunk i visible; ring slot free

        if (i + 1 < CSZ / 16)          // uniform branch: prefetch chunk i+1
            async_ld16(&sA[(i + 1) & 1][srow * ROWP + sseg * 8],
                       Amat + (size_t)((i + 1) * 16 + srow) * DSZ + sseg * 8);

        const _Float16* arp = &sA[i & 1][arow * ROWP + koff];
        v8f acc = {};
        #pragma unroll
        for (int kc = 0; kc < 4; ++kc) {
            v8h lo = *(const v8h*)(arp + kc * 32);
            v8h hi = *(const v8h*)(arp + kc * 32 + 16);
            v16h a = __builtin_shufflevector(lo, hi,
                0, 1, 2, 3, 4, 5, 6, 7, 8, 9, 10, 11, 12, 13, 14, 15);
            acc = __builtin_amdgcn_wmma_f32_16x16x32_f16(
                false, a, false, breg[kc], (short)0, acc, false, false);
        }

        const int c0 = i * 16;
        const int tc = rel - c0;                   // matches when tc == j
        v4f ic0 = *(const v4f*)(&sIC[c0 + sel * 8]);
        v4f ic1 = *(const v4f*)(&sIC[c0 + sel * 8 + 4]);
        #pragma unroll
        for (int j = 0; j < 8; ++j) {
            float logit = acc[j] * (j < 4 ? ic0[j & 3] : ic1[j & 3]);
            sumexp += __expf(logit);               // logits in [-2,2]: safe
            pos    += (tc == j) ? logit : 0.0f;
        }
    }

    // combine the two row-halves of each column
    sumexp += __shfl_xor(sumexp, 16, 32);
    pos    += __shfl_xor(pos,    16, 32);
    float colval = pos - __logf(sumexp);           // pos - lse for this column

    // sum the 16 columns of this wave (values duplicated across halves)
    colval += __shfl_xor(colval, 1, 32);
    colval += __shfl_xor(colval, 2, 32);
    colval += __shfl_xor(colval, 4, 32);
    colval += __shfl_xor(colval, 8, 32);

    if (lane == 0) red[wave] = colval;
    __syncthreads();
    if (tid == 0) {
        float s = 0.0f;
        #pragma unroll
        for (int w = 0; w < 8; ++w) s += red[w];
        partials[blockIdx.x] = s;
    }
}

// ---------------------------------------------------------------------------
// finalize: deterministic tree-reduce of 256 partials, apply scale.
// ---------------------------------------------------------------------------
__global__ __launch_bounds__(256) void pcl_finalize(
    const float* __restrict__ partials,
    const int*   __restrict__ lb,
    float* __restrict__ out)
{
    __shared__ float s[256];
    int t = threadIdx.x;
    s[t] = partials[t];
    __syncthreads();
    for (int off = 128; off > 0; off >>= 1) {
        if (t < off) s[t] += s[t + off];
        __syncthreads();
    }
    if (t == 0)
        out[0] = s[0] * (-(float)lb[0]) * 0.5f / (float)BSZ;
}

// ---------------------------------------------------------------------------
extern "C" void kernel_launch(void* const* d_in, const int* in_sizes, int n_in,
                              void* d_out, int out_size, void* d_ws, size_t ws_size,
                              hipStream_t stream)
{
    const float* features       = (const float*)d_in[0];
    const float* features_I     = (const float*)d_in[1];
    const float* M_kmeans       = (const float*)d_in[2];
    const float* M_kmeans_I     = (const float*)d_in[3];
    const float* concentrations = (const float*)d_in[4];
    const float* concentrationsI= (const float*)d_in[5];
    const int*   labels         = (const int*)d_in[6];
    const int*   labels_I       = (const int*)d_in[7];
    const int*   lb             = (const int*)d_in[8];
    float*       out            = (float*)d_out;

    // workspace layout
    _Float16* M16   = (_Float16*)d_ws;                       // 512 KB
    _Float16* M16I  = M16 + (size_t)CSZ * DSZ;               // 512 KB
    float*    ivc   = (float*)((char*)d_ws + 1048576);       // 8 KB
    float*    ivcI  = ivc + CSZ;                             // 8 KB
    float*    parts = ivcI + CSZ;                            // 1 KB

    pcl_prep<<<1024, 256, 0, stream>>>(M_kmeans, M_kmeans_I,
                                       concentrations, concentrationsI,
                                       M16, M16I, ivc, ivcI);
    pcl_main<<<256, 256, 0, stream>>>(features, features_I, M16, M16I,
                                      ivc, ivcI, labels, labels_I, parts);
    pcl_finalize<<<1, 256, 0, stream>>>(parts, lb, out);
}
